// DSACA_Net_85237920957003
// MI455X (gfx1250) — compile-verified, matrix operand order checked by reading, not run on hardware
//
#include <hip/hip_runtime.h>
#include <cstddef>
#include <cstdint>

// ---------------- problem constants ----------------
#define B_   64
#define C_   2048
#define N_   196
#define A_   312
#define APAD_ 320
#define E_   300
#define S_   150
#define L_   256
#define EPS_ 1e-5f
#define SCALE_ 20.0f

// output layout (floats)
#define OUT_SCORE   0
#define OUT_GF      9600
#define OUT_MFBF    140672
#define OUT_AMAP    4054400
#define OUT_ATTR    7968128
#define OUT_IMGATT  7988096

// ---------------- WMMA vector types ----------------
typedef __attribute__((ext_vector_type(16))) _Float16 v16h;
typedef __attribute__((ext_vector_type(8)))  float    v8f;
typedef __attribute__((ext_vector_type(4)))  float    f32x4;

union FragU { v16h v; f32x4 f[2]; };

#define WMMA_F16(Afrag, Bfrag, Cacc) \
  __builtin_amdgcn_wmma_f32_16x16x32_f16(false, (Afrag), false, (Bfrag), (short)0, (Cacc), false, false)

// Fragment loader from an f16 matrix, rows contiguous along K.
// ISA 16-bit 16x32 layout: lane&15 = row(M)/col(N); lanes>=16 shift K base by 8;
// halves 0..7 = K kb..kb+7, halves 8..15 = K kb+16..kb+23.
__device__ __forceinline__ v16h load_frag_h(const _Float16* base, int ld, int rc0,
                                            int k0, int lane) {
  int m  = lane & 15;
  int kb = k0 + ((lane >> 4) << 3);
  const _Float16* p = base + (size_t)(rc0 + m) * ld + kb;
  FragU u;
  u.f[0] = *(const f32x4*)(p);
  u.f[1] = *(const f32x4*)(p + 16);
  return u.v;
}

// 32x64 output tile per wave, software-pipelined in registers:
// next k-step's 6 fragments are in flight while the current 8 WMMAs issue.
__device__ __forceinline__ void wmma_tile(const _Float16* __restrict__ A, int lda,
                                          int row0,
                                          const _Float16* __restrict__ Wh, int ldw,
                                          int col0, int K, int lane, v8f acc[2][4]) {
  v16h a0 = load_frag_h(A, lda, row0,      0, lane);
  v16h a1 = load_frag_h(A, lda, row0 + 16, 0, lane);
  v16h b0 = load_frag_h(Wh, ldw, col0,      0, lane);
  v16h b1 = load_frag_h(Wh, ldw, col0 + 16, 0, lane);
  v16h b2 = load_frag_h(Wh, ldw, col0 + 32, 0, lane);
  v16h b3 = load_frag_h(Wh, ldw, col0 + 48, 0, lane);
  const _Float16* arow = A + (size_t)(row0 + (lane & 15)) * lda;
  for (int k0 = 32; k0 < K; k0 += 32) {
    __builtin_prefetch(arow + k0 + 32, 0, 3);   // global_prefetch_b8
    v16h na0 = load_frag_h(A, lda, row0,      k0, lane);
    v16h na1 = load_frag_h(A, lda, row0 + 16, k0, lane);
    v16h nb0 = load_frag_h(Wh, ldw, col0,      k0, lane);
    v16h nb1 = load_frag_h(Wh, ldw, col0 + 16, k0, lane);
    v16h nb2 = load_frag_h(Wh, ldw, col0 + 32, k0, lane);
    v16h nb3 = load_frag_h(Wh, ldw, col0 + 48, k0, lane);
    acc[0][0] = WMMA_F16(a0, b0, acc[0][0]);
    acc[0][1] = WMMA_F16(a0, b1, acc[0][1]);
    acc[0][2] = WMMA_F16(a0, b2, acc[0][2]);
    acc[0][3] = WMMA_F16(a0, b3, acc[0][3]);
    acc[1][0] = WMMA_F16(a1, b0, acc[1][0]);
    acc[1][1] = WMMA_F16(a1, b1, acc[1][1]);
    acc[1][2] = WMMA_F16(a1, b2, acc[1][2]);
    acc[1][3] = WMMA_F16(a1, b3, acc[1][3]);
    a0 = na0; a1 = na1; b0 = nb0; b1 = nb1; b2 = nb2; b3 = nb3;
  }
  acc[0][0] = WMMA_F16(a0, b0, acc[0][0]);
  acc[0][1] = WMMA_F16(a0, b1, acc[0][1]);
  acc[0][2] = WMMA_F16(a0, b2, acc[0][2]);
  acc[0][3] = WMMA_F16(a0, b3, acc[0][3]);
  acc[1][0] = WMMA_F16(a1, b0, acc[1][0]);
  acc[1][1] = WMMA_F16(a1, b1, acc[1][1]);
  acc[1][2] = WMMA_F16(a1, b2, acc[1][2]);
  acc[1][3] = WMMA_F16(a1, b3, acc[1][3]);
}

// ---------------- staging kernels ----------------

// x [B,C,N] f32 -> feat_h [B,N,C] f16
__global__ void __launch_bounds__(256)
k_cast_transpose(const float* __restrict__ x, _Float16* __restrict__ out) {
  size_t idx = (size_t)blockIdx.x * blockDim.x + threadIdx.x;
  size_t total = (size_t)B_ * C_ * N_;
  if (idx >= total) return;
  int n = (int)(idx % N_);
  size_t bc = idx / N_;
  int c = (int)(bc % C_);
  int b = (int)(bc / C_);
  out[((size_t)b * N_ + n) * C_ + c] = (_Float16)x[idx];
}

// straight f32 -> f16 cast (weight staging)
__global__ void __launch_bounds__(256)
k_cast_f16(const float* __restrict__ w, _Float16* __restrict__ out, int n) {
  int idx = blockIdx.x * blockDim.x + threadIdx.x;
  if (idx < n) out[idx] = (_Float16)w[idx];
}

// UUt_h[t,l] = UU[l,t] (f16), rows t in [0,320) zero-padded past 312
__global__ void __launch_bounds__(256)
k_uut_h(const float* __restrict__ UU, _Float16* __restrict__ UUt) {
  int idx = blockIdx.x * blockDim.x + threadIdx.x;
  if (idx >= APAD_ * L_) return;
  int t = idx / L_, l = idx % L_;
  UUt[idx] = (t < A_) ? (_Float16)UU[(size_t)l * A_ + t] : (_Float16)0.f;
}

// global_feat[b,c] = mean_n x[b,c,n]  (one wave per (b,c))
__global__ void __launch_bounds__(256)
k_global_mean(const float* __restrict__ x, float* __restrict__ gf) {
  int wid  = blockIdx.x * (blockDim.x >> 5) + (threadIdx.x >> 5);
  int lane = threadIdx.x & 31;
  if (wid >= B_ * C_) return;
  const float* row = x + (size_t)wid * N_;
  float s = 0.f;
  for (int i = lane; i < N_; i += 32) s += row[i];
  for (int off = 16; off; off >>= 1) s += __shfl_xor(s, off, 32);
  if (lane == 0) gf[wid] = s / (float)N_;
}

// ---------------- score path ----------------
__global__ void __launch_bounds__(160)
k_sa_inv(const float* __restrict__ seen, float* __restrict__ sa_inv) {
  int s = blockIdx.x * blockDim.x + threadIdx.x;
  if (s >= S_) return;
  float acc = 0.f;
  const float* r = seen + (size_t)s * A_;
  for (int a = 0; a < A_; ++a) acc += r[a] * r[a];
  sa_inv[s] = 1.f / (sqrtf(acc) + EPS_);
}

__global__ void __launch_bounds__(320)
k_score(const float* __restrict__ gf_all, const float* __restrict__ V,
        const float* __restrict__ seen, const float* __restrict__ sa_inv,
        float* __restrict__ score) {
  __shared__ float gf[C_];
  __shared__ float gsn[A_];
  __shared__ float invb;
  int b = blockIdx.x, t = threadIdx.x;
  const float* g = gf_all + (size_t)b * C_;
  for (int i = t; i < C_; i += blockDim.x) gf[i] = g[i];
  __syncthreads();
  if (t < A_) {
    float acc = 0.f;
    for (int c = 0; c < C_; ++c) acc += gf[c] * V[(size_t)c * A_ + t];
    gsn[t] = acc;
  }
  __syncthreads();
  if (t == 0) {
    float s = 0.f;
    for (int i = 0; i < A_; ++i) s += gsn[i] * gsn[i];
    invb = 1.f / (sqrtf(s) + EPS_);
  }
  __syncthreads();
  if (t < S_) {
    float acc = 0.f;
    const float* sr = seen + (size_t)t * A_;
    for (int a = 0; a < A_; ++a) acc += gsn[a] * sr[a];
    score[(size_t)b * S_ + t] = SCALE_ * acc * invb * sa_inv[t];
  }
}

// ---------------- WMMA GEMM kernels (32x64 tiles) ----------------

// Out[p,o] = sum_k A[p,k]*W[o,k] + bias[o]   (q_conv / k_conv, O multiple of 64)
__global__ void __launch_bounds__(256)
k_gemm_qk(const _Float16* __restrict__ Af, const _Float16* __restrict__ Wh,
          const float* __restrict__ bias, float* __restrict__ Out,
          int rows, int K, int O) {
  int wid  = blockIdx.x * (blockDim.x >> 5) + (threadIdx.x >> 5);
  int lane = threadIdx.x & 31;
  int nstrip = O >> 6;
  int row0 = (wid / nstrip) << 5;
  int col0 = (wid % nstrip) << 6;
  if (row0 >= rows) return;
  v8f acc[2][4] = {};
  wmma_tile(Af, K, row0, Wh, K, col0, K, lane, acc);
  int nl = lane & 15;
  int rbase = ((lane >> 4) << 3);
  for (int i = 0; i < 2; ++i) {
    int mb = row0 + 16 * i + rbase;
    for (int j = 0; j < 4; ++j) {
      int n = col0 + 16 * j + nl;
      float bn = bias[n];
      for (int r = 0; r < 8; ++r)
        Out[(size_t)(mb + r) * O + n] = acc[i][j][r] + bn;
    }
  }
}

// v_conv with fused epilogue: att = gamma*v*diag + feat; writes f32 [B,C,N]
// output plus f16 [B*N, C] staging copy.
__global__ void __launch_bounds__(256)
k_gemm_vconv(const _Float16* __restrict__ Af, const _Float16* __restrict__ Wh,
             const float* __restrict__ bias, const float* __restrict__ gamma,
             const float* __restrict__ diag, const float* __restrict__ x,
             float* __restrict__ out_att, _Float16* __restrict__ att_h,
             int rows, int K) {
  int wid  = blockIdx.x * (blockDim.x >> 5) + (threadIdx.x >> 5);
  int lane = threadIdx.x & 31;
  const int nstrip = C_ >> 6;
  int row0 = (wid / nstrip) << 5;
  int col0 = (wid % nstrip) << 6;
  if (row0 >= rows) return;
  v8f acc[2][4] = {};
  wmma_tile(Af, K, row0, Wh, K, col0, K, lane, acc);
  int nl = lane & 15;
  int rbase = ((lane >> 4) << 3);
  float g = gamma[0];
  for (int i = 0; i < 2; ++i) {
    int mb = row0 + 16 * i + rbase;
    for (int j = 0; j < 4; ++j) {
      int o = col0 + 16 * j + nl;
      float bo = bias[o];
      for (int r = 0; r < 8; ++r) {
        int p = mb + r;
        int b = p / N_, n = p % N_;
        size_t xi = ((size_t)b * C_ + o) * N_ + n;
        float att = g * (acc[i][j][r] + bo) * diag[p] + x[xi];
        out_att[xi] = att;
        att_h[(size_t)p * C_ + o] = (_Float16)att;
      }
    }
  }
}

// img_L = relu(img2L(img_attention)) -> f16 [B*N, 256]
__global__ void __launch_bounds__(256)
k_gemm_imgL(const _Float16* __restrict__ Af, const _Float16* __restrict__ Wh,
            const float* __restrict__ bias, _Float16* __restrict__ OutL,
            int rows, int K, int O) {
  int wid  = blockIdx.x * (blockDim.x >> 5) + (threadIdx.x >> 5);
  int lane = threadIdx.x & 31;
  int nstrip = O >> 6;
  int row0 = (wid / nstrip) << 5;
  int col0 = (wid % nstrip) << 6;
  if (row0 >= rows) return;
  v8f acc[2][4] = {};
  wmma_tile(Af, K, row0, Wh, K, col0, K, lane, acc);
  int nl = lane & 15;
  int rbase = ((lane >> 4) << 3);
  for (int i = 0; i < 2; ++i) {
    int mb = row0 + 16 * i + rbase;
    for (int j = 0; j < 4; ++j) {
      int n = col0 + 16 * j + nl;
      float bn = bias[n];
      for (int r = 0; r < 8; ++r)
        OutL[(size_t)(mb + r) * O + n] = (_Float16)fmaxf(acc[i][j][r] + bn, 0.f);
    }
  }
}

// mfb_F[b,t,n] = (sum_l imgL[p,l] * UUt[t,l]) * mfbQ[n,t]   (UUt zero-padded to 320)
__global__ void __launch_bounds__(256)
k_gemm_mfb(const _Float16* __restrict__ Lh, const _Float16* __restrict__ UUt,
           const float* __restrict__ mfbQ, float* __restrict__ outF,
           int rows, int K) {
  int wid  = blockIdx.x * (blockDim.x >> 5) + (threadIdx.x >> 5);
  int lane = threadIdx.x & 31;
  const int nstrip = APAD_ >> 6;  // 5
  int row0 = (wid / nstrip) << 5;
  int col0 = (wid % nstrip) << 6;
  if (row0 >= rows) return;
  v8f acc[2][4] = {};
  wmma_tile(Lh, K, row0, UUt, K, col0, K, lane, acc);
  int nl = lane & 15;
  int rbase = ((lane >> 4) << 3);
  for (int i = 0; i < 2; ++i) {
    int mb = row0 + 16 * i + rbase;
    for (int j = 0; j < 4; ++j) {
      int t = col0 + 16 * j + nl;
      if (t >= A_) continue;
      for (int r = 0; r < 8; ++r) {
        int p = mb + r;
        int b = p / N_, n = p % N_;
        outF[((size_t)b * A_ + t) * N_ + n] = acc[i][j][r] * mfbQ[(size_t)n * A_ + t];
      }
    }
  }
}

// ---------------- attention diag softmax ----------------
__global__ void __launch_bounds__(256)
k_diag(const float* __restrict__ qf, const float* __restrict__ kf,
       float* __restrict__ diag) {
  __shared__ float q[L_];
  __shared__ float l[N_];
  int p = blockIdx.x;          // pixel = b*N + n
  int b = p / N_, n = p % N_;
  int t = threadIdx.x;
  const float* qr = qf + (size_t)p * L_;
  for (int i = t; i < L_; i += blockDim.x) q[i] = qr[i];
  __syncthreads();
  if (t < N_) {
    const float* kr = kf + ((size_t)b * N_ + t) * L_;
    float acc = 0.f;
    for (int c = 0; c < L_; ++c) acc += q[c] * kr[c];
    l[t] = acc;
  }
  __syncthreads();
  if (t == 0) {
    float mx = -3.4e38f;
    for (int i = 0; i < N_; ++i) mx = fmaxf(mx, l[i]);
    float s = 0.f;
    for (int i = 0; i < N_; ++i) s += __expf(l[i] - mx);
    diag[p] = __expf(l[n] - mx) / s;
  }
}

// ---------------- mfb softmax + mean ----------------
__global__ void __launch_bounds__(256)
k_softmax_map(const float* __restrict__ F, float* __restrict__ amap,
              float* __restrict__ attn) {
  __shared__ float v[N_];
  __shared__ float mx_s, sum_s;
  int row = blockIdx.x;        // b*A + t
  size_t base = (size_t)row * N_;
  int t = threadIdx.x;
  if (t < N_) v[t] = F[base + t];
  __syncthreads();
  if (t == 0) {
    float mx = -3.4e38f, s = 0.f, mean = 0.f;
    for (int i = 0; i < N_; ++i) mx = fmaxf(mx, v[i]);
    for (int i = 0; i < N_; ++i) { s += __expf(v[i] - mx); mean += v[i]; }
    attn[row] = mean / (float)N_;
    mx_s = mx; sum_s = s;
  }
  __syncthreads();
  if (t < N_) amap[base + t] = __expf(v[t] - mx_s) / sum_s;
}

// ---------------- small dense linear (text branch etc.) ----------------
__global__ void __launch_bounds__(256)
k_linear(const float* __restrict__ X, const float* __restrict__ W,
         const float* __restrict__ bias, float* __restrict__ Out,
         int M, int K, int O, int relu) {
  int idx = blockIdx.x * blockDim.x + threadIdx.x;
  int total = M * O;
  for (; idx < total; idx += gridDim.x * blockDim.x) {
    int m = idx / O, o = idx % O;
    float acc = bias ? bias[o] : 0.f;
    const float* xr = X + (size_t)m * K;
    const float* wr = W + (size_t)o * K;
    for (int k = 0; k < K; ++k) acc += xr[k] * wr[k];
    if (relu) acc = fmaxf(acc, 0.f);
    Out[idx] = acc;
  }
}

__global__ void __launch_bounds__(320)
k_text_attn(const float* __restrict__ tq, const float* __restrict__ tk,
            const float* __restrict__ tv, const float* __restrict__ gamma,
            const float* __restrict__ w2v, float* __restrict__ tatt) {
  __shared__ float l[A_];
  __shared__ float mx_s, sum_s;
  int a = blockIdx.x, t = threadIdx.x;
  for (int m = t; m < A_; m += blockDim.x) {
    float acc = 0.f;
    const float* qa = tq + (size_t)a * L_;
    const float* km = tk + (size_t)m * L_;
    for (int c = 0; c < L_; ++c) acc += qa[c] * km[c];
    l[m] = acc;
  }
  __syncthreads();
  if (t == 0) {
    float mx = -3.4e38f, s = 0.f;
    for (int i = 0; i < A_; ++i) mx = fmaxf(mx, l[i]);
    for (int i = 0; i < A_; ++i) s += __expf(l[i] - mx);
    mx_s = mx; sum_s = s;
  }
  __syncthreads();
  for (int m = t; m < A_; m += blockDim.x) l[m] = __expf(l[m] - mx_s) / sum_s;
  __syncthreads();
  float g = gamma[0];
  for (int e = t; e < E_; e += blockDim.x) {
    float acc = 0.f;
    for (int m = 0; m < A_; ++m) acc += l[m] * tv[(size_t)m * E_ + e];
    tatt[(size_t)a * E_ + e] = g * acc + w2v[(size_t)a * E_ + e];
  }
}

// mfbQ[n,t] = sum_l VV[l,n] * tlin[t,l]
__global__ void __launch_bounds__(256)
k_mfbQ(const float* __restrict__ VV, const float* __restrict__ tlin,
       float* __restrict__ mfbQ) {
  int idx = blockIdx.x * blockDim.x + threadIdx.x;
  if (idx >= N_ * A_) return;
  int n = idx / A_, t = idx % A_;
  float acc = 0.f;
  for (int l = 0; l < L_; ++l) acc += VV[(size_t)l * N_ + n] * tlin[(size_t)t * L_ + l];
  mfbQ[idx] = acc;
}

// ---------------- host ----------------
extern "C" void kernel_launch(void* const* d_in, const int* in_sizes, int n_in,
                              void* d_out, int out_size, void* d_ws, size_t ws_size,
                              hipStream_t stream) {
  const float* x    = (const float*)d_in[0];
  const float* w2v  = (const float*)d_in[1];
  const float* seen = (const float*)d_in[2];
  const float* V    = (const float*)d_in[3];
  const float* qcw  = (const float*)d_in[4];
  const float* qcb  = (const float*)d_in[5];
  const float* kcw  = (const float*)d_in[6];
  const float* kcb  = (const float*)d_in[7];
  const float* vcw  = (const float*)d_in[8];
  const float* vcb  = (const float*)d_in[9];
  const float* gimg = (const float*)d_in[10];
  const float* qw   = (const float*)d_in[11];
  const float* qb   = (const float*)d_in[12];
  const float* kw   = (const float*)d_in[13];
  const float* kb   = (const float*)d_in[14];
  const float* vw   = (const float*)d_in[15];
  const float* vb   = (const float*)d_in[16];
  const float* gtxt = (const float*)d_in[17];
  const float* i2Lw = (const float*)d_in[18];
  const float* i2Lb = (const float*)d_in[19];
  const float* t2Lw = (const float*)d_in[20];
  const float* t2Lb = (const float*)d_in[21];
  const float* UU   = (const float*)d_in[22];
  const float* VV   = (const float*)d_in[23];
  const float* a2aw = (const float*)d_in[24];
  const float* a2ab = (const float*)d_in[25];

  float* out        = (float*)d_out;
  float* out_score  = out + OUT_SCORE;
  float* out_gf     = out + OUT_GF;
  float* out_mfbF   = out + OUT_MFBF;
  float* out_amap   = out + OUT_AMAP;
  float* out_attr   = out + OUT_ATTR;
  float* out_imgatt = out + OUT_IMGATT;

  char* ws = (char*)d_ws;
  _Float16* feat_h = (_Float16*)(ws);                  //  51,380,224 B
  _Float16* att_h  = (_Float16*)(ws + 51380224);       //  51,380,224 B
  _Float16* imgL_h = (_Float16*)(ws + 102760448);      //   6,422,528 B
  float*    q_f    = (float*)(ws + 109182976);         //  12,845,056 B
  float*    k_f    = (float*)(ws + 122028032);         //  12,845,056 B
  float*    diag   = (float*)(ws + 134873088);         //     200,704 B
  _Float16* qcw_h  = (_Float16*)(ws + 135073792);      //   1,048,576 B
  _Float16* kcw_h  = (_Float16*)(ws + 136122368);      //   1,048,576 B
  _Float16* vcw_h  = (_Float16*)(ws + 137170944);      //   8,388,608 B
  _Float16* i2Lw_h = (_Float16*)(ws + 145559552);      //   1,048,576 B
  _Float16* UUt_h  = (_Float16*)(ws + 146608128);      //     163,840 B
  float*    tq     = (float*)(ws + 146771968);         //     319,488 B
  float*    tk     = (float*)(ws + 147091456);         //     319,488 B
  float*    tv     = (float*)(ws + 147410944);         //     374,400 B
  float*    tatt   = (float*)(ws + 147785344);         //     374,400 B
  float*    tlin   = (float*)(ws + 148159744);         //     319,488 B
  float*    mfbQ   = (float*)(ws + 148479232);         //     244,608 B
  float*    attn   = (float*)(ws + 148723840);         //      79,872 B
  float*    sa_inv = (float*)(ws + 148803712);         //         640 B

  const int rows = B_ * N_;            // 12544 pixel rows, /32 = 392 tiles

  // -------- staging --------
  {
    size_t total = (size_t)B_ * C_ * N_;
    k_cast_transpose<<<(int)((total + 255) / 256), 256, 0, stream>>>(x, feat_h);
  }
  k_cast_f16<<<(L_ * C_ + 255) / 256, 256, 0, stream>>>(qcw, qcw_h, L_ * C_);
  k_cast_f16<<<(L_ * C_ + 255) / 256, 256, 0, stream>>>(kcw, kcw_h, L_ * C_);
  k_cast_f16<<<(C_ * C_ + 255) / 256, 256, 0, stream>>>(vcw, vcw_h, C_ * C_);
  k_cast_f16<<<(L_ * C_ + 255) / 256, 256, 0, stream>>>(i2Lw, i2Lw_h, L_ * C_);
  k_uut_h<<<(APAD_ * L_ + 255) / 256, 256, 0, stream>>>(UU, UUt_h);
  k_global_mean<<<(B_ * C_) / 8, 256, 0, stream>>>(x, out_gf);

  // -------- score path --------
  k_sa_inv<<<1, 160, 0, stream>>>(seen, sa_inv);
  k_score<<<B_, 320, 0, stream>>>(out_gf, V, seen, sa_inv, out_score);

  // -------- image attention --------
  {
    int waves = (rows / 32) * (L_ / 64);   // 1568
    k_gemm_qk<<<waves / 8, 256, 0, stream>>>(feat_h, qcw_h, qcb, q_f, rows, C_, L_);
    k_gemm_qk<<<waves / 8, 256, 0, stream>>>(feat_h, kcw_h, kcb, k_f, rows, C_, L_);
  }
  k_diag<<<rows, 256, 0, stream>>>(q_f, k_f, diag);
  {
    int waves = (rows / 32) * (C_ / 64);   // 12544
    k_gemm_vconv<<<waves / 8, 256, 0, stream>>>(feat_h, vcw_h, vcb, gimg, diag, x,
                                                out_imgatt, att_h, rows, C_);
  }

  // -------- text branch (tiny) --------
  k_linear<<<(A_ * L_ + 255) / 256, 256, 0, stream>>>(w2v, qw, qb, tq, A_, E_, L_, 0);
  k_linear<<<(A_ * L_ + 255) / 256, 256, 0, stream>>>(w2v, kw, kb, tk, A_, E_, L_, 0);
  k_linear<<<(A_ * E_ + 255) / 256, 256, 0, stream>>>(w2v, vw, vb, tv, A_, E_, E_, 0);
  k_text_attn<<<A_, 320, 0, stream>>>(tq, tk, tv, gtxt, w2v, tatt);
  k_linear<<<(A_ * L_ + 255) / 256, 256, 0, stream>>>(tatt, t2Lw, t2Lb, tlin, A_, E_, L_, 1);
  k_mfbQ<<<(N_ * A_ + 255) / 256, 256, 0, stream>>>(VV, tlin, mfbQ);

  // -------- MFB fusion --------
  {
    int waves = (rows / 32) * (L_ / 64);   // 1568
    k_gemm_imgL<<<waves / 8, 256, 0, stream>>>(att_h, i2Lw_h, i2Lb, imgL_h, rows, C_, L_);
  }
  {
    int waves = (rows / 32) * (APAD_ / 64);  // 1960
    k_gemm_mfb<<<waves / 8, 256, 0, stream>>>(imgL_h, UUt_h, mfbQ, out_mfbF, rows, L_);
  }
  k_softmax_map<<<B_ * A_, 256, 0, stream>>>(out_mfbF, out_amap, attn);
  k_linear<<<(B_ * A_ + 255) / 256, 256, 0, stream>>>(attn, a2aw, a2ab, out_attr,
                                                      B_, A_, A_, 0);
}
